// CORAL_model_82300163326640
// MI455X (gfx1250) — matrix operand order repaired and checked by reading, not compiled, not run on hardware
//
#include <hip/hip_runtime.h>
#include <cstdint>
#include <cstddef>

// ---------------------------------------------------------------------------
// Problem constants (from reference)
// ---------------------------------------------------------------------------
constexpr int NN  = 20000;     // cells
constexpr int EE  = 320000;    // edges
constexpr int M2T = EE + NN;   // edges + self loops = 340000
constexpr int DVV = 1000;
constexpr int DCC = 40;
constexpr int CTT = 20;
constexpr int LL  = 64;
constexpr int VDD = 32;
constexpr int NSS = 512;
constexpr float EPSF    = 1e-10f;
constexpr float NB_EPSF = 1e-5f;
constexpr float MEAN_NUM = (float)NN / (float)NSS;   // counts.mean() == N/NS exactly

// ---------------------------------------------------------------------------
// Input pointer index map (insertion order of setup_inputs / params dicts)
// ---------------------------------------------------------------------------
enum {
  IN_X = 0, IN_CT = 1,
  P_ENCV1_W = 2, P_ENCV1_B, P_ENCV2_W, P_ENCV2_B,
  P_ENCC1_W, P_ENCC1_B, P_ENCC2_W, P_ENCC2_B,
  P_ATTNQ_W, P_ATTNQ_B, P_ATTNK_W, P_ATTNK_B, P_ATTNV_W, P_ATTNV_B,
  P_DEC1_W, P_DEC1_B, P_DEC2_W, P_DEC2_B,
  P_GAT1_W, P_GAT1_AS, P_GAT1_AD, P_GAT1_B,
  P_GAT2_W, P_GAT2_AS, P_GAT2_AD, P_GAT2_B,
  P_HID_W, P_HID_B, P_VSCALE_W, P_VSCALE_B, P_CSCALE_W, P_CSCALE_B,
  P_CTDEC_W, P_CTDEC_B, P_VM_W, P_VM_B, P_VLV_W, P_VLV_B,
  P_PX_R, P_PX_R_SC, P_PY_R,
  IN_EDGE, IN_SPOT, IN_NSPOTS
};

// ---------------------------------------------------------------------------
// Output offsets (floats), tuple return order of reference
// ---------------------------------------------------------------------------
constexpr size_t O_vrate  = 0;
constexpr size_t O_agg    = O_vrate  + (size_t)NN * DVV;
constexpr size_t O_pxr    = O_agg    + (size_t)NSS * DVV;
constexpr size_t O_crate  = O_pxr    + DVV;
constexpr size_t O_pyr    = O_crate  + (size_t)NN * DCC;
constexpr size_t O_combmu = O_pyr    + DCC;
constexpr size_t O_comblv = O_combmu + (size_t)NN * LL;
constexpr size_t O_z2     = O_comblv + (size_t)NN * LL;
constexpr size_t O_genct  = O_z2     + (size_t)NN * LL;
constexpr size_t O_pxrsc  = O_genct  + (size_t)NN * CTT;
constexpr size_t O_deconv = O_pxrsc  + DVV;
constexpr size_t O_qxi    = O_deconv + (size_t)NN * DVV;
constexpr size_t O_v      = O_qxi    + (size_t)NN * DVV;
constexpr size_t O_vm     = O_v      + (size_t)NN * VDD;
constexpr size_t O_vlv    = O_vm     + (size_t)NN * VDD;
constexpr size_t O_a1     = O_vlv    + (size_t)NN * VDD;
constexpr size_t O_a2     = O_a1     + (size_t)M2T * 4;

// ---------------------------------------------------------------------------
// Workspace offsets (floats)
// ---------------------------------------------------------------------------
constexpr size_t W_buf1020 = 0;                                  // dec_in then enc_v_in
constexpr size_t W_dh      = W_buf1020 + (size_t)NN * 1020;      // N x 128 hidden
constexpr size_t W_lin     = W_dh      + (size_t)NN * 128;       // N x 1000 pre-softmax
constexpr size_t W_vlib    = W_lin     + (size_t)NN * 1000;
constexpr size_t W_clib    = W_vlib    + NN;
constexpr size_t W_counts  = W_clib    + NN;
constexpr size_t W_encc    = W_counts  + NSS;                    // N x 40
constexpr size_t W_vh      = W_encc    + (size_t)NN * 40;        // N x 128
constexpr size_t W_ch      = W_vh      + (size_t)NN * 128;       // N x 128
constexpr size_t W_h1      = W_ch      + (size_t)NN * 128;       // N x 256 (gat features)
constexpr size_t W_z1      = W_h1      + (size_t)NN * 256;       // N x 256 (gat layer1 out)
constexpr size_t W_ssum    = W_z1      + (size_t)NN * 256;       // N x 4
constexpr size_t W_dsum    = W_ssum    + (size_t)NN * 4;
constexpr size_t W_m       = W_dsum    + (size_t)NN * 4;
constexpr size_t W_s       = W_m       + (size_t)NN * 4;
constexpr size_t W_w       = W_s       + (size_t)NN * 4;         // M2 x 4 edge weights
constexpr size_t W_gv      = W_w       + (size_t)M2T * 4;        // v_mu_g  (N x 64)
constexpr size_t W_gc      = W_gv      + (size_t)NN * 64;        // c_mu_g  (N x 64)
constexpr size_t W_sum64   = W_gc      + (size_t)NN * 64;        // (a+b)   (N x 64)
constexpr size_t W_z       = W_sum64   + (size_t)NN * 64;        // z       (N x 64)
constexpr size_t W_zc      = W_z       + (size_t)NN * 64;        // N x 84
constexpr size_t W_hid     = W_zc      + (size_t)NN * 84;        // N x 128

// ---------------------------------------------------------------------------
// WMMA GEMM:  C[M,Nc] = act(A[M,K](lda) * B[K,Nc](ldb) + bias)
// act: 0 = +bias, 1 = relu(+bias), 2 = 0.5*acc + bias
// block = 128 threads (4 waves), 64x64 tile, each wave 32x32 (2x2 WMMA)
// A tile in LDS row-major [row][k] (stride 36 halves, 8B aligned rows)
// B tile in LDS col-major [col][k] (stride 36 halves) so each lane's
// fragment is two runs of 8 contiguous halves -> ds_load_b64s.
// ---------------------------------------------------------------------------
typedef __attribute__((ext_vector_type(16))) _Float16 v16h;
typedef __attribute__((ext_vector_type(8)))  float    v8f;
typedef __attribute__((ext_vector_type(4)))  _Float16 h4v;

constexpr int GSA  = 36;   // halves per A row
constexpr int GSBc = 36;   // halves per B col

__global__ __launch_bounds__(128)
void gemm_wmma_kernel(const float* __restrict__ A, int lda,
                      const float* __restrict__ B, int ldb,
                      const float* __restrict__ bias,
                      float* __restrict__ C, int ldc,
                      int M, int Nc, int K, int act)
{
  __shared__ __align__(16) _Float16 As[64 * GSA];
  __shared__ __align__(16) _Float16 Bs[64 * GSBc];

  const int tid  = threadIdx.x;
  const int lane = tid & 31;
  const int wave = tid >> 5;
  const int wm   = (wave >> 1) * 32;
  const int wn   = (wave & 1) * 32;
  const int rowBase = blockIdx.y * 64;
  const int colBase = blockIdx.x * 64;
  const bool fullM = (rowBase + 64 <= M);
  const bool fullN = (colBase + 64 <= Nc);

  v8f acc[2][2];
  #pragma unroll
  for (int i = 0; i < 2; ++i)
    #pragma unroll
    for (int j = 0; j < 2; ++j)
      #pragma unroll
      for (int e = 0; e < 8; ++e) acc[i][j][e] = 0.0f;

  const int half_id = lane >> 4;
  const int l15     = lane & 15;

  for (int k0 = 0; k0 < K; k0 += 32) {
    const bool fullK = (k0 + 32 <= K);

    // ---- stage A tile (64 rows x 32 k) --------------------------------
    if (fullM && fullK) {
      #pragma unroll
      for (int t = 0; t < 4; ++t) {
        int idx = tid + t * 128;             // 0..511 float4s
        int r = idx >> 3, kc = (idx & 7) << 2;
        const float4 v = *(const float4*)(A + (size_t)(rowBase + r) * lda + (k0 + kc));
        h4v h = { (_Float16)v.x, (_Float16)v.y, (_Float16)v.z, (_Float16)v.w };
        *(h4v*)(As + r * GSA + kc) = h;
      }
    } else {
      for (int idx = tid; idx < 64 * 32; idx += 128) {
        int r = idx >> 5, kk = idx & 31;
        int gr = rowBase + r, gk = k0 + kk;
        float v = (gr < M && gk < K) ? A[(size_t)gr * lda + gk] : 0.0f;
        As[r * GSA + kk] = (_Float16)v;
      }
    }
    // ---- stage B tile (32 k x 64 cols), transposed into LDS -----------
    if (fullN && fullK) {
      #pragma unroll
      for (int t = 0; t < 4; ++t) {
        int idx = tid + t * 128;             // 0..511 float4s
        int kk = idx >> 4, cc = (idx & 15) << 2;
        const float4 v = *(const float4*)(B + (size_t)(k0 + kk) * ldb + (colBase + cc));
        Bs[(cc + 0) * GSBc + kk] = (_Float16)v.x;
        Bs[(cc + 1) * GSBc + kk] = (_Float16)v.y;
        Bs[(cc + 2) * GSBc + kk] = (_Float16)v.z;
        Bs[(cc + 3) * GSBc + kk] = (_Float16)v.w;
      }
    } else {
      for (int idx = tid; idx < 64 * 32; idx += 128) {
        int kk = idx >> 6, c = idx & 63;
        int gk = k0 + kk, gc = colBase + c;
        float v = (gk < K && gc < Nc) ? B[(size_t)gk * ldb + gc] : 0.0f;
        Bs[c * GSBc + kk] = (_Float16)v;
      }
    }

    // ---- prefetch next K tile into caches while the barrier drains ----
    if (k0 + 32 < K) {
      if (tid < 64) {
        int gr = rowBase + tid; if (gr >= M) gr = M - 1;
        __builtin_prefetch(A + (size_t)gr * lda + (k0 + 32), 0, 1);
      } else if (tid < 96) {
        int gk = k0 + 32 + (tid - 64);
        if (gk < K) __builtin_prefetch(B + (size_t)gk * ldb + colBase, 0, 1);
      }
    }
    __syncthreads();

    // ---- fragments ----------------------------------------------------
    // A: lane<16 holds K 0..7,16..23 of row l15; lane>=16 holds K 8..15,24..31
    v16h afrag[2], bfrag[2];
    #pragma unroll
    for (int i = 0; i < 2; ++i) {
      const _Float16* ap = As + (wm + i * 16 + l15) * GSA + half_id * 8;
      h4v a0 = *(const h4v*)(ap);
      h4v a1 = *(const h4v*)(ap + 4);
      h4v a2 = *(const h4v*)(ap + 16);
      h4v a3 = *(const h4v*)(ap + 20);
      v16h a;
      #pragma unroll
      for (int j = 0; j < 4; ++j) {
        a[j] = a0[j]; a[4 + j] = a1[j]; a[8 + j] = a2[j]; a[12 + j] = a3[j];
      }
      afrag[i] = a;
    }
    // B: lane<16 holds K 0..7,16..23 of col l15; lane>=16 holds K 8..15,24..31
    #pragma unroll
    for (int j2 = 0; j2 < 2; ++j2) {
      const _Float16* bp = Bs + (wn + j2 * 16 + l15) * GSBc + half_id * 8;
      h4v b0 = *(const h4v*)(bp);
      h4v b1 = *(const h4v*)(bp + 4);
      h4v b2 = *(const h4v*)(bp + 16);
      h4v b3 = *(const h4v*)(bp + 20);
      v16h b;
      #pragma unroll
      for (int j = 0; j < 4; ++j) {
        b[j] = b0[j]; b[4 + j] = b1[j]; b[8 + j] = b2[j]; b[12 + j] = b3[j];
      }
      bfrag[j2] = b;
    }
    #pragma unroll
    for (int i = 0; i < 2; ++i)
      #pragma unroll
      for (int j = 0; j < 2; ++j)
        acc[i][j] = __builtin_amdgcn_wmma_f32_16x16x32_f16(
            false, afrag[i], false, bfrag[j], (short)0, acc[i][j], false, false);
    __syncthreads();
  }

  // ---- store C: VGPR e holds rows e and 8+e (half-wave split), col = l15 --
  if (fullM && fullN) {
    #pragma unroll
    for (int i = 0; i < 2; ++i) {
      #pragma unroll
      for (int j = 0; j < 2; ++j) {
        int cc = colBase + wn + j * 16 + l15;
        float bv = bias ? bias[cc] : 0.0f;
        #pragma unroll
        for (int e = 0; e < 8; ++e) {
          int rr = rowBase + wm + i * 16 + half_id * 8 + e;
          float v = acc[i][j][e];
          if (act == 2) v = 0.5f * v + bv; else v += bv;
          if (act == 1) v = fmaxf(v, 0.0f);
          C[(size_t)rr * ldc + cc] = v;
        }
      }
    }
  } else {
    for (int i = 0; i < 2; ++i) {
      for (int j = 0; j < 2; ++j) {
        int cc = colBase + wn + j * 16 + l15;
        if (cc >= Nc) continue;
        float bv = bias ? bias[cc] : 0.0f;
        for (int e = 0; e < 8; ++e) {
          int rr = rowBase + wm + i * 16 + half_id * 8 + e;
          if (rr >= M) continue;
          float v = acc[i][j][e];
          if (act == 2) v = 0.5f * v + bv; else v += bv;
          if (act == 1) v = fmaxf(v, 0.0f);
          C[(size_t)rr * ldc + cc] = v;
        }
      }
    }
  }
}

// ---------------------------------------------------------------------------
// RNG helpers (deterministic counter hash, graph-capture safe)
// ---------------------------------------------------------------------------
__device__ inline uint32_t pcg_next(uint32_t& state) {
  state = state * 747796405u + 2891336453u;
  uint32_t w = ((state >> ((state >> 28u) + 4u)) ^ state) * 277803737u;
  return (w >> 22u) ^ w;
}
__device__ inline float u01(uint32_t h) {
  return (float)(h >> 8) * (1.0f / 16777216.0f) + (0.5f / 16777216.0f);
}
__device__ inline float rng_normal(uint32_t& st) {
  float u1 = u01(pcg_next(st)), u2 = u01(pcg_next(st));
  return sqrtf(-2.0f * logf(u1)) * cosf(6.28318530718f * u2);
}
__device__ inline float softplusf(float x) {
  return fmaxf(x, 0.0f) + log1pf(expf(-fabsf(x)));
}
__device__ inline float rng_gamma(float a, uint32_t& st) {
  float boost = 1.0f;
  if (a < 1.0f) { float u = u01(pcg_next(st)); boost = powf(u, 1.0f / a); a += 1.0f; }
  float d = a - (1.0f / 3.0f);
  float c = rsqrtf(9.0f * d);
  for (int it = 0; it < 16; ++it) {
    float xn = rng_normal(st);
    float t = 1.0f + c * xn;
    if (t <= 0.0f) continue;
    float v = t * t * t;
    float u = u01(pcg_next(st));
    if (logf(u) < 0.5f * xn * xn + d - d * v + d * logf(v)) return boost * d * v;
  }
  return boost * d;
}
__device__ inline float rng_poisson(float lam, uint32_t& st) {
  if (!(lam > 0.0f)) return 0.0f;
  if (lam < 12.0f) {
    float Lth = expf(-lam), p = 1.0f; int k = 0;
    for (int it = 0; it < 64; ++it) {
      p *= u01(pcg_next(st));
      if (p <= Lth) break;
      ++k;
    }
    return (float)k;
  }
  float x = lam + sqrtf(lam) * rng_normal(st);
  return fmaxf(0.0f, floorf(x + 0.5f));
}
__device__ inline float atomicMaxF(float* addr, float val) {
  int* ai = (int*)addr;
  int old = *ai;
  while (__int_as_float(old) < val) {
    int assumed = old;
    old = atomicCAS(ai, assumed, __float_as_int(val));
    if (old == assumed) break;
  }
  return __int_as_float(old);
}

// ---------------------------------------------------------------------------
// Support kernels
// ---------------------------------------------------------------------------
__global__ void zero_kernel(float* p, size_t n) {
  size_t i = (size_t)blockIdx.x * blockDim.x + threadIdx.x;
  if (i < n) p[i] = 0.0f;
}

// per-row: vlib/clib, dec_in = [log1p(vx), ct], encc_in = log1p(cx)
__global__ void prep_kernel(const float* __restrict__ x, const float* __restrict__ ct,
                            float* __restrict__ dec_in, float* __restrict__ encc_in,
                            float* __restrict__ vlib, float* __restrict__ clib) {
  int n = blockIdx.x;
  const float* row = x + (size_t)n * (DVV + DCC);
  __shared__ float sv[256], sc[256];
  float av = 0.0f, ac = 0.0f;
  for (int g = threadIdx.x; g < DVV; g += 256) {
    float v = row[g]; av += v;
    dec_in[(size_t)n * 1020 + g] = log1pf(v);
  }
  for (int g = threadIdx.x; g < DCC; g += 256) {
    float v = row[DVV + g]; ac += v;
    encc_in[(size_t)n * DCC + g] = log1pf(v);
  }
  for (int g = threadIdx.x; g < CTT; g += 256)
    dec_in[(size_t)n * 1020 + DVV + g] = ct[(size_t)n * CTT + g];
  sv[threadIdx.x] = av; sc[threadIdx.x] = ac;
  __syncthreads();
  for (int off = 128; off > 0; off >>= 1) {
    if (threadIdx.x < off) { sv[threadIdx.x] += sv[threadIdx.x + off]; sc[threadIdx.x] += sc[threadIdx.x + off]; }
    __syncthreads();
  }
  if (threadIdx.x == 0) { vlib[n] = log1pf(sv[0]); clib[n] = log1pf(sc[0]); }
}

// row softmax; optional *exp(logscale[row]) + eps_add
__global__ void softmax_kernel(const float* __restrict__ in, float* __restrict__ out,
                               int W, const float* __restrict__ logscale, float eps_add) {
  int n = blockIdx.x;
  const float* r = in + (size_t)n * W;
  float* o = out + (size_t)n * W;
  __shared__ float sh[256];
  int t = threadIdx.x;
  float m = -3.4e38f;
  for (int g = t; g < W; g += 256) m = fmaxf(m, r[g]);
  sh[t] = m; __syncthreads();
  for (int off = 128; off > 0; off >>= 1) { if (t < off) sh[t] = fmaxf(sh[t], sh[t + off]); __syncthreads(); }
  m = sh[0]; __syncthreads();
  float s = 0.0f;
  for (int g = t; g < W; g += 256) s += expf(r[g] - m);
  sh[t] = s; __syncthreads();
  for (int off = 128; off > 0; off >>= 1) { if (t < off) sh[t] += sh[t + off]; __syncthreads(); }
  float inv  = 1.0f / sh[0];
  float mult = logscale ? expf(logscale[n]) : 1.0f;
  for (int g = t; g < W; g += 256) o[g] = expf(r[g] - m) * inv * mult + eps_add;
}

__global__ void counts_kernel(const int* __restrict__ spot, float* __restrict__ counts) {
  int i = blockIdx.x * blockDim.x + threadIdx.x;
  if (i < NN) atomicAdd(&counts[spot[i]], 1.0f);
}

// NB sampling -> q_xi and enc_v input (log1p(q_xi) ++ cell_type)
__global__ void sample_qxi_kernel(const float* __restrict__ deconv, const float* __restrict__ vlib,
                                  const float* __restrict__ px_r_sc, float* __restrict__ qxi,
                                  float* __restrict__ encv_in, const float* __restrict__ ct) {
  int n = blockIdx.x;
  float libf = expf(vlib[n] * (1.0f / MEAN_NUM));
  for (int g = threadIdx.x; g < DVV; g += 256) {
    float mu    = libf * deconv[(size_t)n * DVV + g] + EPSF;
    float theta = expf(softplusf(px_r_sc[g]) + EPSF);
    float conc  = theta + NB_EPSF;
    uint32_t st = ((uint32_t)n * 1000003u + (uint32_t)g * 7919u) ^ 0xBEEF1234u;
    float gam = rng_gamma(conc, st);
    float lam = gam * (mu + NB_EPSF) / conc;
    float q   = rng_poisson(lam, st);
    qxi[(size_t)n * DVV + g]      = q;
    encv_in[(size_t)n * 1020 + g] = log1pf(q);
  }
  for (int g = threadIdx.x; g < CTT; g += 256)
    encv_in[(size_t)n * 1020 + DVV + g] = ct[(size_t)n * CTT + g];
}

// GAT helpers -----------------------------------------------------------------
__device__ inline void edge_sd(const int* __restrict__ ei, int i, int& s, int& d) {
  if (i < EE) { s = ei[i]; d = ei[EE + i]; }
  else        { s = i - EE; d = i - EE; }
}

__global__ void attn_sums_kernel(const float* __restrict__ h, const float* __restrict__ a_s,
                                 const float* __restrict__ a_d, float* __restrict__ ssum,
                                 float* __restrict__ dsum, int H, int C) {
  int idx = blockIdx.x * blockDim.x + threadIdx.x;
  if (idx >= NN * H) return;
  int n = idx / H, hh = idx % H;
  const float* hr = h + ((size_t)n * H + hh) * C;
  const float* as = a_s + (size_t)hh * C;
  const float* ad = a_d + (size_t)hh * C;
  float s1 = 0.0f, s2 = 0.0f;
  for (int c = 0; c < C; ++c) { float v = hr[c]; s1 += v * as[c]; s2 += v * ad[c]; }
  ssum[idx] = s1; dsum[idx] = s2;
}

__global__ void init_ms_kernel(float* m, float* s, int n) {
  int i = blockIdx.x * blockDim.x + threadIdx.x;
  if (i < n) { m[i] = -3.4e38f; s[i] = 0.0f; }
}

__global__ void edge_max_kernel(const float* __restrict__ ssum, const float* __restrict__ dsum,
                                const int* __restrict__ ei, float* __restrict__ m, int H) {
  int idx = blockIdx.x * blockDim.x + threadIdx.x;
  if (idx >= M2T * H) return;
  int i = idx / H, hh = idx % H;
  int s, d; edge_sd(ei, i, s, d);
  float e = ssum[s * H + hh] + dsum[d * H + hh];
  e = e > 0.0f ? e : 0.2f * e;
  atomicMaxF(&m[d * H + hh], e);
}

__global__ void edge_w_kernel(const float* __restrict__ ssum, const float* __restrict__ dsum,
                              const int* __restrict__ ei, const float* __restrict__ m,
                              float* __restrict__ ws, float* __restrict__ w, int H) {
  int idx = blockIdx.x * blockDim.x + threadIdx.x;
  if (idx >= M2T * H) return;
  int i = idx / H, hh = idx % H;
  int s, d; edge_sd(ei, i, s, d);
  float e = ssum[s * H + hh] + dsum[d * H + hh];
  e = e > 0.0f ? e : 0.2f * e;
  float wv = expf(e - m[d * H + hh]);
  w[idx] = wv;
  atomicAdd(&ws[d * H + hh], wv);
}

__global__ void edge_agg_kernel(const float* __restrict__ hfeat, const float* __restrict__ w,
                                const float* __restrict__ ws, const int* __restrict__ ei,
                                float* __restrict__ out, float* __restrict__ alpha_out,
                                int H, int C) {
  int idx = blockIdx.x * blockDim.x + threadIdx.x;
  if (idx >= M2T * H) return;
  int i = idx / H, hh = idx % H;
  int s, d; edge_sd(ei, i, s, d);
  float alpha = w[idx] / (ws[d * H + hh] + 1e-16f);
  if (alpha_out) alpha_out[idx] = alpha;
  const float* hr = hfeat + ((size_t)s * H + hh) * C;
  float* orow = out + ((size_t)d * H + hh) * C;
  for (int c = 0; c < C; ++c) atomicAdd(&orow[c], alpha * hr[c]);
}

__global__ void bias_act_kernel(float* __restrict__ out, const float* __restrict__ b,
                                int D, int relu) {
  size_t i = (size_t)blockIdx.x * blockDim.x + threadIdx.x;
  if (i >= (size_t)NN * D) return;
  float v = out[i] + b[i % D];
  if (relu) v = fmaxf(v, 0.0f);
  out[i] = v;
}

__global__ void add2_strided_kernel(const float* __restrict__ a, int lda,
                                    const float* __restrict__ b, int ldb,
                                    float* __restrict__ o, int ldo, int D) {
  size_t i = (size_t)blockIdx.x * blockDim.x + threadIdx.x;
  if (i >= (size_t)NN * D) return;
  size_t n = i / D, c = i % D;
  o[n * ldo + c] = a[n * lda + c] + b[n * ldb + c];
}

__global__ void sample_z_kernel(const float* __restrict__ mu, const float* __restrict__ lv,
                                float* __restrict__ z, int D, uint32_t salt) {
  size_t i = (size_t)blockIdx.x * blockDim.x + threadIdx.x;
  if (i >= (size_t)NN * D) return;
  uint32_t st = ((uint32_t)i * 2654435761u) ^ salt;
  z[i] = mu[i] + rng_normal(st) * expf(0.5f * lv[i]);
}

__global__ void concat_zc_kernel(const float* __restrict__ z, const float* __restrict__ ct,
                                 float* __restrict__ zc) {
  size_t i = (size_t)blockIdx.x * blockDim.x + threadIdx.x;
  if (i >= (size_t)NN * (LL + CTT)) return;
  size_t n = i / (LL + CTT), c = i % (LL + CTT);
  zc[i] = (c < LL) ? z[n * LL + c] : ct[n * CTT + (c - LL)];
}

__global__ void agg_add_kernel(const float* __restrict__ vrate, const int* __restrict__ spot,
                               float* __restrict__ agg) {
  int n = blockIdx.x;
  int sidx = spot[n];
  for (int g = threadIdx.x; g < DVV; g += 256)
    atomicAdd(&agg[(size_t)sidx * DVV + g], vrate[(size_t)n * DVV + g]);
}

__global__ void agg_div_kernel(float* __restrict__ agg, const float* __restrict__ counts) {
  size_t i = (size_t)blockIdx.x * blockDim.x + threadIdx.x;
  if (i >= (size_t)NSS * DVV) return;
  agg[i] /= fmaxf(counts[i / DVV], 1.0f);
}

__global__ void softplus_kernel(const float* __restrict__ in, float* __restrict__ out, int n) {
  int i = blockIdx.x * blockDim.x + threadIdx.x;
  if (i < n) out[i] = softplusf(in[i]) + EPSF;
}

// ---------------------------------------------------------------------------
// Host-side helpers
// ---------------------------------------------------------------------------
static inline int cdiv_l(long a, int b) { return (int)((a + b - 1) / b); }

static void gemm_launch(const float* A, int lda, const float* B, int ldb,
                        const float* bias, float* C, int ldc,
                        int M, int Nc, int K, int act, hipStream_t st) {
  dim3 grid((Nc + 63) / 64, (M + 63) / 64);
  gemm_wmma_kernel<<<grid, dim3(128), 0, st>>>(A, lda, B, ldb, bias, C, ldc, M, Nc, K, act);
}

struct GatBufs { float *h1, *z1, *ssum, *dsum, *m, *s, *w; };

static void run_gat_stack(const float* xin, int ld_in,
                          float* out64, float* a1_out, float* a2_out,
                          const float* g1W, const float* g1as, const float* g1ad, const float* g1b,
                          const float* g2W, const float* g2as, const float* g2ad, const float* g2b,
                          const int* ei, const GatBufs& B, hipStream_t st) {
  // ---- layer 1: Din=64 -> H=4, C=64 (out 256), relu
  gemm_launch(xin, ld_in, g1W, 256, nullptr, B.h1, 256, NN, 256, 64, 0, st);
  {
    const int H = 4, C = 64;
    attn_sums_kernel<<<cdiv_l((long)NN * H, 256), 256, 0, st>>>(B.h1, g1as, g1ad, B.ssum, B.dsum, H, C);
    init_ms_kernel<<<cdiv_l((long)NN * H, 256), 256, 0, st>>>(B.m, B.s, NN * H);
    edge_max_kernel<<<cdiv_l((long)M2T * H, 256), 256, 0, st>>>(B.ssum, B.dsum, ei, B.m, H);
    edge_w_kernel<<<cdiv_l((long)M2T * H, 256), 256, 0, st>>>(B.ssum, B.dsum, ei, B.m, B.s, B.w, H);
    zero_kernel<<<cdiv_l((long)NN * 256, 256), 256, 0, st>>>(B.z1, (size_t)NN * 256);
    edge_agg_kernel<<<cdiv_l((long)M2T * H, 256), 256, 0, st>>>(B.h1, B.w, B.s, ei, B.z1, a1_out, H, C);
    bias_act_kernel<<<cdiv_l((long)NN * 256, 256), 256, 0, st>>>(B.z1, g1b, 256, 1);
  }
  // ---- layer 2: Din=256 -> H=1, C=64
  gemm_launch(B.z1, 256, g2W, 64, nullptr, B.h1, 64, NN, 64, 256, 0, st);
  {
    const int H = 1, C = 64;
    attn_sums_kernel<<<cdiv_l((long)NN, 256), 256, 0, st>>>(B.h1, g2as, g2ad, B.ssum, B.dsum, H, C);
    init_ms_kernel<<<cdiv_l((long)NN, 256), 256, 0, st>>>(B.m, B.s, NN);
    edge_max_kernel<<<cdiv_l((long)M2T, 256), 256, 0, st>>>(B.ssum, B.dsum, ei, B.m, H);
    edge_w_kernel<<<cdiv_l((long)M2T, 256), 256, 0, st>>>(B.ssum, B.dsum, ei, B.m, B.s, B.w, H);
    zero_kernel<<<cdiv_l((long)NN * 64, 256), 256, 0, st>>>(out64, (size_t)NN * 64);
    edge_agg_kernel<<<cdiv_l((long)M2T, 256), 256, 0, st>>>(B.h1, B.w, B.s, ei, out64, a2_out, H, C);
    bias_act_kernel<<<cdiv_l((long)NN * 64, 256), 256, 0, st>>>(out64, g2b, 64, 0);
  }
}

// ---------------------------------------------------------------------------
// kernel_launch
// ---------------------------------------------------------------------------
extern "C" void kernel_launch(void* const* d_in, const int* in_sizes, int n_in,
                              void* d_out, int out_size, void* d_ws, size_t ws_size,
                              hipStream_t stream) {
  (void)in_sizes; (void)n_in; (void)out_size; (void)ws_size;
  auto F  = [&](int i) { return (const float*)d_in[i]; };
  const float* x     = F(IN_X);
  const float* ct    = F(IN_CT);
  const int* edge    = (const int*)d_in[IN_EDGE];
  const int* spot    = (const int*)d_in[IN_SPOT];

  float* ws  = (float*)d_ws;
  float* out = (float*)d_out;

  GatBufs GB{ws + W_h1, ws + W_z1, ws + W_ssum, ws + W_dsum, ws + W_m, ws + W_s, ws + W_w};

  // 1. prep: libs + dec input + enc_c input
  prep_kernel<<<NN, 256, 0, stream>>>(x, ct, ws + W_buf1020, ws + W_encc, ws + W_vlib, ws + W_clib);

  // 2. deconvolution branch
  gemm_launch(ws + W_buf1020, 1020, F(P_DEC1_W), 128, F(P_DEC1_B), ws + W_dh, 128, NN, 128, 1020, 1, stream);
  gemm_launch(ws + W_dh, 128, F(P_DEC2_W), 1000, F(P_DEC2_B), ws + W_lin, 1000, NN, 1000, 128, 0, stream);
  softmax_kernel<<<NN, 256, 0, stream>>>(ws + W_lin, out + O_deconv, DVV, nullptr, 0.0f);

  // 3. spot counts
  zero_kernel<<<cdiv_l(NSS, 256), 256, 0, stream>>>(ws + W_counts, (size_t)NSS);
  counts_kernel<<<cdiv_l(NN, 256), 256, 0, stream>>>(spot, ws + W_counts);

  // 4. NB sampling -> q_xi + enc_v input (overwrites dec input buffer)
  sample_qxi_kernel<<<NN, 256, 0, stream>>>(out + O_deconv, ws + W_vlib, F(P_PX_R_SC),
                                            out + O_qxi, ws + W_buf1020, ct);

  // 5. encoders
  gemm_launch(ws + W_buf1020, 1020, F(P_ENCV1_W), 128, F(P_ENCV1_B), ws + W_dh, 128, NN, 128, 1020, 1, stream);
  gemm_launch(ws + W_dh, 128, F(P_ENCV2_W), 128, F(P_ENCV2_B), ws + W_vh, 128, NN, 128, 128, 0, stream);
  gemm_launch(ws + W_encc, 40, F(P_ENCC1_W), 128, F(P_ENCC1_B), ws + W_dh, 128, NN, 128, 40, 1, stream);
  gemm_launch(ws + W_dh, 128, F(P_ENCC2_W), 128, F(P_ENCC2_B), ws + W_ch, 128, NN, 128, 128, 0, stream);

  // 6. GAT stacks on v_mu (vh[:, :64]) and c_mu (ch[:, :64])
  run_gat_stack(ws + W_vh, 128, ws + W_gv, nullptr, nullptr,
                F(P_GAT1_W), F(P_GAT1_AS), F(P_GAT1_AD), F(P_GAT1_B),
                F(P_GAT2_W), F(P_GAT2_AS), F(P_GAT2_AD), F(P_GAT2_B), edge, GB, stream);
  run_gat_stack(ws + W_ch, 128, ws + W_gc, nullptr, nullptr,
                F(P_GAT1_W), F(P_GAT1_AS), F(P_GAT1_AD), F(P_GAT1_B),
                F(P_GAT2_W), F(P_GAT2_AS), F(P_GAT2_AD), F(P_GAT2_B), edge, GB, stream);

  // 7. comb_mu = ((v_mu_g + c_mu_g) @ Vw)/2 + Vb ; comb_lv likewise from lv slices
  add2_strided_kernel<<<cdiv_l((long)NN * 64, 256), 256, 0, stream>>>(ws + W_gv, 64, ws + W_gc, 64, ws + W_sum64, 64, 64);
  gemm_launch(ws + W_sum64, 64, F(P_ATTNV_W), 64, F(P_ATTNV_B), out + O_combmu, 64, NN, 64, 64, 2, stream);
  add2_strided_kernel<<<cdiv_l((long)NN * 64, 256), 256, 0, stream>>>(ws + W_vh + 64, 128, ws + W_ch + 64, 128, ws + W_sum64, 64, 64);
  gemm_launch(ws + W_sum64, 64, F(P_ATTNV_W), 64, F(P_ATTNV_B), out + O_comblv, 64, NN, 64, 64, 2, stream);

  // 8. reparameterize z
  sample_z_kernel<<<cdiv_l((long)NN * 64, 256), 256, 0, stream>>>(out + O_combmu, out + O_comblv, ws + W_z, 64, 0x5A17F00Du);

  // 9. v head
  concat_zc_kernel<<<cdiv_l((long)NN * (LL + CTT), 256), 256, 0, stream>>>(ws + W_z, ct, ws + W_zc);
  gemm_launch(ws + W_zc, 84, F(P_VM_W), 32, F(P_VM_B), out + O_vm, 32, NN, 32, 84, 0, stream);
  gemm_launch(ws + W_zc, 84, F(P_VLV_W), 32, F(P_VLV_B), out + O_vlv, 32, NN, 32, 84, 0, stream);
  sample_z_kernel<<<cdiv_l((long)NN * 32, 256), 256, 0, stream>>>(out + O_vm, out + O_vlv, out + O_v, 32, 0xC0FFEE11u);

  // 10. decoder GAT stack on z (produces z2, a1, a2)
  run_gat_stack(ws + W_z, 64, out + O_z2, out + O_a1, out + O_a2,
                F(P_GAT1_W), F(P_GAT1_AS), F(P_GAT1_AD), F(P_GAT1_B),
                F(P_GAT2_W), F(P_GAT2_AS), F(P_GAT2_AD), F(P_GAT2_B), edge, GB, stream);

  // 11. decoder heads
  gemm_launch(out + O_z2, 64, F(P_HID_W), 128, F(P_HID_B), ws + W_hid, 128, NN, 128, 64, 1, stream);
  gemm_launch(ws + W_hid, 128, F(P_VSCALE_W), 1000, F(P_VSCALE_B), ws + W_lin, 1000, NN, 1000, 128, 0, stream);
  softmax_kernel<<<NN, 256, 0, stream>>>(ws + W_lin, out + O_vrate, DVV, ws + W_vlib, EPSF);
  gemm_launch(ws + W_hid, 128, F(P_CSCALE_W), 40, F(P_CSCALE_B), ws + W_lin, 40, NN, 40, 128, 0, stream);
  softmax_kernel<<<NN, 256, 0, stream>>>(ws + W_lin, out + O_crate, DCC, ws + W_clib, EPSF);
  gemm_launch(out + O_z2, 64, F(P_CTDEC_W), 20, F(P_CTDEC_B), out + O_genct, 20, NN, 20, 64, 0, stream);

  // 12. spot aggregation
  zero_kernel<<<cdiv_l((long)NSS * DVV, 256), 256, 0, stream>>>(out + O_agg, (size_t)NSS * DVV);
  agg_add_kernel<<<NN, 256, 0, stream>>>(out + O_vrate, spot, out + O_agg);
  agg_div_kernel<<<cdiv_l((long)NSS * DVV, 256), 256, 0, stream>>>(out + O_agg, ws + W_counts);

  // 13. dispersion outputs
  softplus_kernel<<<cdiv_l(DVV, 256), 256, 0, stream>>>(F(P_PX_R),    out + O_pxr,   DVV);
  softplus_kernel<<<cdiv_l(DCC, 256), 256, 0, stream>>>(F(P_PY_R),    out + O_pyr,   DCC);
  softplus_kernel<<<cdiv_l(DVV, 256), 256, 0, stream>>>(F(P_PX_R_SC), out + O_pxrsc, DVV);
}
